// NeuroBLASTAttention_32040456029019
// MI455X (gfx1250) — matrix operand order repaired and checked by reading, not compiled, not run on hardware
//
#include <hip/hip_runtime.h>
#include <hip/hip_bf16.h>
#include <math.h>

typedef __bf16 bf16;
typedef __attribute__((ext_vector_type(16))) __bf16 v16bf;
typedef __attribute__((ext_vector_type(8)))  __bf16 v8bf;
typedef __attribute__((ext_vector_type(8)))  float  v8f;
typedef int v4i_ __attribute__((vector_size(16)));

#define B_   2
#define S_   2048
#define HID_ 2048
#define NH_  16
#define NKV_ 4
#define DH_  128
#define EPS_ 1e-6f

// ---- CDNA5 async global->LDS copy (ASYNCcnt path), with sync fallback -----
#if defined(__has_builtin)
#if __has_builtin(__builtin_amdgcn_global_load_async_to_lds_b128)
#define ASYNC_LDS 1
#endif
#endif
#ifndef ASYNC_LDS
#define ASYNC_LDS 0
#endif

__device__ __forceinline__ void cp16(bf16* lds, const bf16* g) {
#if ASYNC_LDS
    __builtin_amdgcn_global_load_async_to_lds_b128(
        (__attribute__((address_space(1))) v4i_*)(g),
        (__attribute__((address_space(3))) v4i_*)(lds), 0, 0);
#else
    *(uint4*)lds = *(const uint4*)g;
#endif
}

__device__ __forceinline__ void async_wait0() {
#if ASYNC_LDS
#if __has_builtin(__builtin_amdgcn_s_wait_asynccnt)
    __builtin_amdgcn_s_wait_asynccnt(0);
#else
    asm volatile("s_wait_asynccnt 0x0" ::: "memory");
#endif
#endif
}

// ---------------------------------------------------------------------------
// One-time fp32 -> bf16 elementwise conversion (8 elements / thread).
// ---------------------------------------------------------------------------
__global__ __launch_bounds__(256)
void cvt_bf16(const float* __restrict__ in, bf16* __restrict__ out)
{
    size_t i = ((size_t)blockIdx.x * 256 + threadIdx.x) * 8;
    float4 a = *(const float4*)(in + i);
    float4 b = *(const float4*)(in + i + 4);
    v8bf o;
    o[0] = (bf16)a.x; o[1] = (bf16)a.y; o[2] = (bf16)a.z; o[3] = (bf16)a.w;
    o[4] = (bf16)b.x; o[5] = (bf16)b.y; o[6] = (bf16)b.z; o[7] = (bf16)b.w;
    *(v8bf*)(out + i) = o;
}

// ---------------------------------------------------------------------------
// One-time fp32 [K,N] -> bf16 transposed [N,K] (32x32 LDS tile transpose).
// ---------------------------------------------------------------------------
__global__ __launch_bounds__(256)
void transpose_cvt(const float* __restrict__ W, bf16* __restrict__ WT,
                   int K, int N)
{
    __shared__ float t[32][33];
    const int bk = blockIdx.y * 32, bn = blockIdx.x * 32;
    const int tx = threadIdx.x & 31, ty = threadIdx.x >> 5;
    #pragma unroll
    for (int r = ty; r < 32; r += 8)
        t[r][tx] = W[(size_t)(bk + r) * N + bn + tx];
    __syncthreads();
    #pragma unroll
    for (int r = ty; r < 32; r += 8)
        WT[(size_t)(bn + r) * K + bk + tx] = (bf16)t[tx][r];
}

// ---------------------------------------------------------------------------
// GEMM: C[M,N] = A[M,K] @ B[K,N] with B given transposed (BT[N,K]).
// All-bf16 operands, f32 WMMA accumulation. Block tile 128x64, K-step 64,
// double-buffered async LDS staging. 256 thr = 8 waves (4Mx2N), wave 32x32.
// Assumes M%128==0, N%64==0, K%64==0 (true for all calls here).
// ---------------------------------------------------------------------------
template <typename OutT>
__global__ __launch_bounds__(256)
void gemm_bf16(const bf16* __restrict__ A, const bf16* __restrict__ BT,
               OutT* __restrict__ C, int M, int N, int K)
{
    __shared__ bf16 As[2][128][72];   // 128x64 (+16B pad)
    __shared__ bf16 Bs[2][64][72];    // 64x64  (+16B pad)

    const int tid  = threadIdx.x;
    const int lane = tid & 31;
    const int wave = tid >> 5;
    const int wm   = wave >> 1;          // 0..3
    const int wn   = wave & 1;           // 0..1
    const int bM   = blockIdx.y * 128;
    const int bN   = blockIdx.x * 64;
    const int hi16 = (lane >= 16) ? 1 : 0;

    auto stage = [&](int buf, int kt) {
        // A tile: 128 rows x 64 cols = 1024 x 16B chunks -> 4 per thread
        #pragma unroll
        for (int i = 0; i < 4; ++i) {
            int id = tid + i * 256;
            int r  = id >> 3;
            int c  = (id & 7) * 8;
            cp16(&As[buf][r][c], A + (size_t)(bM + r) * K + kt + c);
        }
        // BT tile: 64 rows x 64 cols = 512 x 16B chunks -> 2 per thread
        #pragma unroll
        for (int i = 0; i < 2; ++i) {
            int id = tid + i * 256;
            int r  = id >> 3;
            int c  = (id & 7) * 8;
            cp16(&Bs[buf][r][c], BT + (size_t)(bN + r) * K + kt + c);
        }
    };

    v8f acc[2][2];
    acc[0][0] = {}; acc[0][1] = {}; acc[1][0] = {}; acc[1][1] = {};

    stage(0, 0);
    async_wait0();
    __syncthreads();

    const int nstages = K / 64;
    for (int s = 0; s < nstages; ++s) {
        const int cur = s & 1;
        if (s + 1 < nstages) stage(cur ^ 1, (s + 1) * 64);  // prefetch next

        #pragma unroll
        for (int ks = 0; ks < 64; ks += 32) {
            v16bf a[2], b[2];
            #pragma unroll
            for (int i = 0; i < 2; ++i) {
                int m = wm * 32 + i * 16 + (lane & 15);
                #pragma unroll
                for (int j = 0; j < 16; ++j) {
                    int kk = ks + hi16 * 16 + j;
                    a[i][j] = As[cur][m][kk];
                }
            }
            #pragma unroll
            for (int j2 = 0; j2 < 2; ++j2) {
                int n = wn * 32 + j2 * 16 + (lane & 15);
                #pragma unroll
                for (int j = 0; j < 16; ++j)
                    b[j2][j] = Bs[cur][n][ks + hi16 * 16 + j];
            }
            #pragma unroll
            for (int i = 0; i < 2; ++i)
                #pragma unroll
                for (int j2 = 0; j2 < 2; ++j2)
                    acc[i][j2] = __builtin_amdgcn_wmma_f32_16x16x32_bf16(
                        false, a[i], false, b[j2], (short)0, acc[i][j2],
                        false, false);
        }
        async_wait0();
        __syncthreads();
    }

    // C layout: element r -> M = r + (lane>=16 ? 8 : 0), N = lane&15
    #pragma unroll
    for (int i = 0; i < 2; ++i)
        #pragma unroll
        for (int j2 = 0; j2 < 2; ++j2)
            #pragma unroll
            for (int r = 0; r < 8; ++r) {
                int row = bM + wm * 32 + i * 16 + r + hi16 * 8;
                int col = bN + wn * 32 + j2 * 16 + (lane & 15);
                C[(size_t)row * N + col] = (OutT)acc[i][j2][r];
            }
}

// ---------------------------------------------------------------------------
// Fused per-head RMSNorm + RoPE, in place on bf16 (math in f32).
// One wave per (token, head) row of DH=128; lane holds d, d+32, d+64, d+96
// so both rotate_half pairs stay in-lane.
// ---------------------------------------------------------------------------
__global__ __launch_bounds__(256)
void rmsnorm_rope(bf16* __restrict__ X, const float* __restrict__ cosT,
                  const float* __restrict__ sinT, const float* __restrict__ w,
                  int nheads)
{
    const int lane = threadIdx.x & 31;
    const int wave = threadIdx.x >> 5;
    const int row  = blockIdx.x * 8 + wave;   // over B*S*nheads
    const int s    = (row / nheads) % S_;

    bf16* x = X + (size_t)row * DH_;
    const int d0 = lane;
    float x0 = (float)x[d0],      x1 = (float)x[d0 + 32];
    float x2 = (float)x[d0 + 64], x3 = (float)x[d0 + 96];

    float ss = x0 * x0 + x1 * x1 + x2 * x2 + x3 * x3;
    #pragma unroll
    for (int m = 16; m >= 1; m >>= 1) ss += __shfl_xor(ss, m, 32);
    float rn = rsqrtf(ss * (1.0f / DH_) + EPS_);

    x0 *= rn * w[d0];      x1 *= rn * w[d0 + 32];
    x2 *= rn * w[d0 + 64]; x3 *= rn * w[d0 + 96];

    const float* cs = cosT + (size_t)s * DH_;
    const float* sn = sinT + (size_t)s * DH_;
    x[d0]      = (bf16)(x0 * cs[d0]      - x2 * sn[d0]);
    x[d0 + 32] = (bf16)(x1 * cs[d0 + 32] - x3 * sn[d0 + 32]);
    x[d0 + 64] = (bf16)(x2 * cs[d0 + 64] + x0 * sn[d0 + 64]);
    x[d0 + 96] = (bf16)(x3 * cs[d0 + 96] + x1 * sn[d0 + 96]);
}

// ---------------------------------------------------------------------------
// Flash attention (causal, GQA 4:1), bf16 in / bf16 out, f32 accumulation.
// Block = 64 q-rows of one (b,h); 4 waves x 16 rows. Key tile = 32.
// K staged async to LDS (row-major); V staged transposed (VtT[dh][kpos]) so
// the P@V B-fragments are contiguous strips.
// ---------------------------------------------------------------------------
__global__ __launch_bounds__(128)
void flash_attn(const bf16* __restrict__ Q, const bf16* __restrict__ Kb,
                const bf16* __restrict__ Vb, bf16* __restrict__ O)
{
    __shared__ bf16 Kt[32][DH_ + 8];
    __shared__ bf16 VtT[DH_][40];
    __shared__ bf16 Ps[4][16][32];

    const int tid   = threadIdx.x;
    const int lane  = tid & 31;
    const int wave  = tid >> 5;            // 0..3
    const int bh    = blockIdx.y;
    const int b     = bh / NH_;
    const int h     = bh % NH_;
    const int g     = h / (NH_ / NKV_);    // kv head
    const int qbase = blockIdx.x * 64;
    const int hi16  = (lane >= 16) ? 1 : 0;
    const float scale = 0.08838834764831845f;   // 1/sqrt(128)

    // Preload Q A-fragments (4 chunks of K=32 over DH=128) via 16B loads
    v16bf qa[4];
    {
        const int qrow = qbase + wave * 16 + (lane & 15);
        const bf16* qp = Q + ((size_t)(b * S_ + qrow) * NH_ + h) * DH_;
        #pragma unroll
        for (int c = 0; c < 4; ++c) {
            v8bf lo = *(const v8bf*)(qp + c * 32 + hi16 * 8);
            v8bf hi = *(const v8bf*)(qp + c * 32 + 16 + hi16 * 8);
            #pragma unroll
            for (int e = 0; e < 8; ++e) { qa[c][e] = lo[e]; qa[c][8 + e] = hi[e]; }
        }
    }

    v8f oacc[8];
    #pragma unroll
    for (int dc = 0; dc < 8; ++dc) oacc[dc] = {};
    float mrow[8], lrow[8];
    #pragma unroll
    for (int r = 0; r < 8; ++r) { mrow[r] = -1e30f; lrow[r] = 0.0f; }

    const int numT = qbase / 32 + 2;       // causal: tiles up to qbase+63
    for (int t = 0; t < numT; ++t) {
        // K tile 32x128: 512 x 16B chunks -> 4 async copies per thread
        #pragma unroll
        for (int i = 0; i < 4; ++i) {
            int id = tid + i * 128;
            int kr = id >> 4;
            int c  = (id & 15) * 8;
            int kpos = t * 32 + kr;
            cp16(&Kt[kr][c],
                 Kb + ((size_t)(b * S_ + kpos) * NKV_ + g) * DH_ + c);
        }
        // V tile 32x128, transposed into VtT[dh][kpos] (sync scatter)
        #pragma unroll
        for (int i = 0; i < 4; ++i) {
            int id = tid + i * 128;
            int kr = id >> 4;
            int c  = (id & 15) * 8;
            int kpos = t * 32 + kr;
            v8bf vv = *(const v8bf*)(Vb +
                       ((size_t)(b * S_ + kpos) * NKV_ + g) * DH_ + c);
            #pragma unroll
            for (int e = 0; e < 8; ++e) VtT[c + e][kr] = vv[e];
        }
        async_wait0();
        __syncthreads();

        // Skip tiles fully above this wave's diagonal
        const bool active = (t * 32 <= qbase + wave * 16 + 15);
        if (active) {
            float sc[2][8];
            #pragma unroll
            for (int n = 0; n < 2; ++n) {
                v8f s = {};
                #pragma unroll
                for (int c = 0; c < 4; ++c) {
                    v16bf bk;
                    #pragma unroll
                    for (int j = 0; j < 16; ++j)
                        bk[j] = Kt[n * 16 + (lane & 15)][c * 32 + hi16 * 8 + j + ((j < 8) ? 0 : 8)];
                    s = __builtin_amdgcn_wmma_f32_16x16x32_bf16(
                        false, qa[c], false, bk, (short)0, s, false, false);
                }
                int kposg = t * 32 + n * 16 + (lane & 15);
                #pragma unroll
                for (int r = 0; r < 8; ++r) {
                    int qp = qbase + wave * 16 + r + hi16 * 8;
                    sc[n][r] = (kposg > qp) ? -1e9f : s[r] * scale;
                }
            }
            // Online softmax; rows live in 16-lane halves of the wave.
            float pn[2][8];
            #pragma unroll
            for (int r = 0; r < 8; ++r) {
                float mx = fmaxf(sc[0][r], sc[1][r]);
                #pragma unroll
                for (int m = 8; m >= 1; m >>= 1)
                    mx = fmaxf(mx, __shfl_xor(mx, m, 32));
                float mnew = fmaxf(mrow[r], mx);
                float corr = __expf(mrow[r] - mnew);
                float p0 = __expf(sc[0][r] - mnew);
                float p1 = __expf(sc[1][r] - mnew);
                float rs = p0 + p1;
                #pragma unroll
                for (int m = 8; m >= 1; m >>= 1)
                    rs += __shfl_xor(rs, m, 32);
                lrow[r] = lrow[r] * corr + rs;
                mrow[r] = mnew;
                pn[0][r] = p0; pn[1][r] = p1;
                #pragma unroll
                for (int dc = 0; dc < 8; ++dc) oacc[dc][r] *= corr;
            }
            // Re-layout P (C layout -> A layout) via this wave's LDS slab.
            #pragma unroll
            for (int n = 0; n < 2; ++n)
                #pragma unroll
                for (int r = 0; r < 8; ++r)
                    Ps[wave][r + hi16 * 8][n * 16 + (lane & 15)] = (bf16)pn[n][r];
            v16bf pa;
            #pragma unroll
            for (int j = 0; j < 16; ++j) {
                int kk = j + hi16 * 16;
                pa[j] = Ps[wave][lane & 15][kk];
            }
            // P(16x32) @ V(32x128): one WMMA per 16-wide dh chunk
            #pragma unroll
            for (int dc = 0; dc < 8; ++dc) {
                v16bf bv;
                #pragma unroll
                for (int j = 0; j < 16; ++j)
                    bv[j] = VtT[dc * 16 + (lane & 15)][hi16 * 16 + j];
                oacc[dc] = __builtin_amdgcn_wmma_f32_16x16x32_bf16(
                    false, pa, false, bv, (short)0, oacc[dc], false, false);
            }
        }
        __syncthreads();
    }

    // Normalize and write out [B,S,NH,DH] bf16
    #pragma unroll
    for (int r = 0; r < 8; ++r) {
        float inv = 1.0f / lrow[r];
        int row = qbase + wave * 16 + r + hi16 * 8;
        #pragma unroll
        for (int dc = 0; dc < 8; ++dc) {
            int dd = dc * 16 + (lane & 15);
            O[((size_t)(b * S_ + row) * NH_ + h) * DH_ + dd] =
                (bf16)(oacc[dc][r] * inv);
        }
    }
}

// ---------------------------------------------------------------------------
extern "C" void kernel_launch(void* const* d_in, const int* in_sizes, int n_in,
                              void* d_out, int out_size, void* d_ws, size_t ws_size,
                              hipStream_t stream)
{
    (void)in_sizes; (void)n_in; (void)out_size; (void)ws_size;
    const float* hs   = (const float*)d_in[0];
    // d_in[1] = attention_mask: pure causal -1e9 mask, applied analytically.
    const float* cosT = (const float*)d_in[2];
    const float* sinT = (const float*)d_in[3];
    const float* Wq   = (const float*)d_in[4];
    const float* Wk   = (const float*)d_in[5];
    const float* Wv   = (const float*)d_in[6];
    const float* Wo   = (const float*)d_in[7];
    const float* qw   = (const float*)d_in[8];
    const float* kw   = (const float*)d_in[9];
    float* out = (float*)d_out;

    const int M  = B_ * S_;                // 4096
    const int NQ = NH_ * DH_;              // 2048
    const int NK = NKV_ * DH_;             // 512

    bf16* p   = (bf16*)d_ws;
    bf16* hsb = p;  p += (size_t)M * HID_;         // 8M
    bf16* WqT = p;  p += (size_t)NQ * HID_;        // 4M   [N,K]
    bf16* WkT = p;  p += (size_t)NK * HID_;        // 1M
    bf16* WvT = p;  p += (size_t)NK * HID_;        // 1M
    bf16* WoT = p;  p += (size_t)HID_ * NQ;        // 4M
    bf16* q   = p;  p += (size_t)M * NQ;           // 8M
    bf16* k   = p;  p += (size_t)M * NK;           // 2M
    bf16* v   = p;  p += (size_t)M * NK;           // 2M
    bf16* ao  = p;                                 // 8M

    // One-time conversions / weight transposes (fp32 -> bf16)
    cvt_bf16<<<dim3(((size_t)M * HID_) / 2048), dim3(256), 0, stream>>>(hs, hsb);
    transpose_cvt<<<dim3(NQ / 32, HID_ / 32), dim3(256), 0, stream>>>(Wq, WqT, HID_, NQ);
    transpose_cvt<<<dim3(NK / 32, HID_ / 32), dim3(256), 0, stream>>>(Wk, WkT, HID_, NK);
    transpose_cvt<<<dim3(NK / 32, HID_ / 32), dim3(256), 0, stream>>>(Wv, WvT, HID_, NK);
    transpose_cvt<<<dim3(HID_ / 32, NQ / 32), dim3(256), 0, stream>>>(Wo, WoT, NQ, HID_);

    // QKV projections (bf16 WMMA GEMMs, async double-buffered)
    gemm_bf16<bf16><<<dim3(NQ / 64, M / 128), dim3(256), 0, stream>>>(hsb, WqT, q, M, NQ, HID_);
    gemm_bf16<bf16><<<dim3(NK / 64, M / 128), dim3(256), 0, stream>>>(hsb, WkT, k, M, NK, HID_);
    gemm_bf16<bf16><<<dim3(NK / 64, M / 128), dim3(256), 0, stream>>>(hsb, WvT, v, M, NK, HID_);

    // Per-head RMSNorm + RoPE (in place, bf16)
    rmsnorm_rope<<<dim3((M * NH_) / 8), dim3(256), 0, stream>>>(q, cosT, sinT, qw, NH_);
    rmsnorm_rope<<<dim3((M * NKV_) / 8), dim3(256), 0, stream>>>(k, cosT, sinT, kw, NKV_);

    // Causal GQA flash attention
    flash_attn<<<dim3(S_ / 64, B_ * NH_), dim3(128), 0, stream>>>(q, k, v, ao);

    // Output projection (f32 output)
    gemm_bf16<float><<<dim3(HID_ / 64, M / 128), dim3(256), 0, stream>>>(ao, WoT, out, M, HID_, HID_);
}
